// Block_24601572671925
// MI455X (gfx1250) — matrix operand order repaired and checked by reading, not compiled, not run on hardware
//
#include <hip/hip_runtime.h>
#include <math.h>

// ---------------------------------------------------------------------------
// CDNA5 (gfx1250) wave32 WMMA types
// ---------------------------------------------------------------------------
typedef __attribute__((ext_vector_type(16))) __bf16 v16bf;
typedef __attribute__((ext_vector_type(8)))  float  v8f;
typedef __attribute__((ext_vector_type(4)))  unsigned int v4u;
typedef __attribute__((ext_vector_type(4)))  int    v4i;
typedef __attribute__((ext_vector_type(8)))  int    v8i;

#if __has_builtin(__builtin_amdgcn_tensor_load_to_lds)
#define HAVE_TDM 1
#else
#define HAVE_TDM 0
#endif

__device__ inline v8f vzero8() {
  v8f z;
#pragma unroll
  for (int i = 0; i < 8; ++i) z[i] = 0.0f;
  return z;
}

__device__ inline v8f wmma_bf16(v16bf a, v16bf b, v8f c) {
  // (neg_a, A, neg_b, B, c_mod, C, reuse_a, reuse_b)
  return __builtin_amdgcn_wmma_f32_16x16x32_bf16(false, a, false, b, (short)0, c,
                                                 false, false);
}

// A-fragment (16x32, bf16): lane covers row (lane&15); elem e -> k = half*16 + e
__device__ inline v16bf load_a_frag(const float* __restrict__ rowp, int half, float scl) {
  const float* p = rowp + half * 16;
  v16bf f;
#pragma unroll
  for (int i = 0; i < 8; ++i) f[i] = (__bf16)(p[i] * scl);
#pragma unroll
  for (int i = 0; i < 8; ++i) f[i + 8] = (__bf16)(p[i + 8] * scl);
  return f;
}

// B-fragment (32x16, bf16) with B[k][n] = W[n][k] (row n of W contiguous in k):
// lane covers col (lane&15); elem e -> k = half*16 + e  => 16 contiguous floats.
__device__ inline v16bf load_b_fragT(const float* __restrict__ rowp, int half) {
  const float* p = rowp + half * 16;
  v16bf f;
#pragma unroll
  for (int i = 0; i < 16; ++i) f[i] = (__bf16)p[i];
  return f;
}

// B-fragment with B[k][n] = V[k][n] (k-major, stride ld between k's).
__device__ inline v16bf load_b_frag_strided(const float* __restrict__ colp, size_t ld, int half) {
  const float* p = colp + (size_t)(half * 16) * ld;
  v16bf f;
#pragma unroll
  for (int i = 0; i < 16; ++i) f[i] = (__bf16)p[(size_t)i * ld];
  return f;
}

__device__ inline float gelu_exact(float v) {
  return 0.5f * v * (1.0f + erff(v * 0.70710678118654752f));
}

#if HAVE_TDM
// Tensor Data Mover: 2D fp32 tile (tile_h rows x tile_w elems, row stride in
// elems) from global -> LDS (contiguous). D# packed per CDNA5 ISA.
__device__ inline void tdm_load_2d_f32(void* lds_dst, const float* gsrc,
                                       int tile_w, int tile_h, int row_stride_elems) {
  unsigned lds_addr = (unsigned)(size_t)lds_dst;           // generic low 32 = LDS byte addr
  unsigned long long ga = (unsigned long long)(size_t)gsrc;
  v4u g0;
  g0[0] = 1u;                                              // count=1, user mode, no gather
  g0[1] = lds_addr;                                        // lds_addr [63:32]
  g0[2] = (unsigned)(ga & 0xffffffffu);                    // global_addr[31:0]
  g0[3] = (unsigned)((ga >> 32) & 0x1ffffffu) | (2u << 30);// addr[56:32] | type=2
  v8i g1;
  g1[0] = (int)(2u << 16);                    // data_size=4B; wg_mask=0; no flags
  g1[1] = (int)((unsigned)tile_w << 16);      // tensor_dim0[15:0] @ bits 63:48
  g1[2] = (int)((unsigned)tile_h << 16);      // tensor_dim1[15:0] @ bits 95:80
  g1[3] = (int)((unsigned)tile_w << 16);      // tile_dim0 @ bits 127:112
  g1[4] = (int)(unsigned)tile_h;              // tile_dim1 @ bits 143:128 (tile_dim2=0)
  g1[5] = row_stride_elems;                   // tensor_dim0_stride[31:0]
  g1[6] = 0;                                  // stride[47:32]=0, dim1_stride lo=0
  g1[7] = 0;
  v4i z4 = {0, 0, 0, 0};
#if defined(__clang_major__) && (__clang_major__ >= 23)
  v8i z8 = {0, 0, 0, 0, 0, 0, 0, 0};
  __builtin_amdgcn_tensor_load_to_lds(g0, g1, z4, z4, z8, 0);
#else
  __builtin_amdgcn_tensor_load_to_lds(g0, g1, z4, z4, 0);
#endif
}
#endif

// ---------------------------------------------------------------------------
// Generic bf16-WMMA GEMM: C[M,N] = act(A[M,K] @ W[N,K]^T + bias)
// M%64==0, N%(NT*16)==0, K%32==0. Block = 128 thr (4 waves); wave computes a
// 16 x (NT*16) tile; grid = (N/(NT*16), M/64). act: 0=none, 1=exact GELU.
// ---------------------------------------------------------------------------
template <int NT>
__global__ __launch_bounds__(128) void gemm_bf16_kernel(
    const float* __restrict__ A, const float* __restrict__ W,
    const float* __restrict__ bias, float* __restrict__ C,
    int M, int N, int K, int act) {
  const int lane = threadIdx.x & 31;
  const int wave = threadIdx.x >> 5;
  const int half = lane >> 4;
  const int l15  = lane & 15;

  const int m0 = blockIdx.y * 64 + wave * 16;
  const int n0 = blockIdx.x * (NT * 16);
  const int mrow = m0 + l15;

  v8f acc[NT];
#pragma unroll
  for (int nt = 0; nt < NT; ++nt) acc[nt] = vzero8();

  const float* arow = A + (size_t)mrow * K;
  for (int k = 0; k < K; k += 32) {
    v16bf af = load_a_frag(arow + k, half, 1.0f);
#pragma unroll
    for (int nt = 0; nt < NT; ++nt) {
      v16bf bf_ = load_b_fragT(W + (size_t)(n0 + nt * 16 + l15) * K + k, half);
      acc[nt] = wmma_bf16(af, bf_, acc[nt]);
    }
  }

#pragma unroll
  for (int nt = 0; nt < NT; ++nt) {
    const int n = n0 + nt * 16 + l15;
    const float bb = bias ? bias[n] : 0.0f;
#pragma unroll
    for (int r = 0; r < 8; ++r) {
      float v = acc[nt][r] + bb;
      if (act == 1) v = gelu_exact(v);
      C[(size_t)(m0 + r + 8 * half) * N + n] = v;
    }
  }
}

// ---------------------------------------------------------------------------
// Attention for (B=8, N=1024, H=12, hd=64), qkv laid out (B,N,3,H,hd): row
// stride 2304, q at +0, k at +768, v at +1536. One wave per (b,h,qtile16).
// Scores staged in 64KB LDS, in-LDS softmax; AV consumes V tiles streamed
// into LDS by the Tensor Data Mover (double-buffered, TENSORcnt-synced).
// ---------------------------------------------------------------------------
#define NTOK 1024
#define DMODEL 768
#define C3 2304
#define HD 64

__global__ __launch_bounds__(32) void attn_kernel(const float* __restrict__ qkv,
                                                  float* __restrict__ out) {
  __shared__ float S[16 * NTOK];        // 64 KB score rows
#if HAVE_TDM
  __shared__ float Vt[2][32 * HD];      // 2 x 8 KB V tiles (double buffer)
#endif

  const int lane = threadIdx.x;
  const int half = lane >> 4;
  const int l15  = lane & 15;
  const int qt = blockIdx.x;   // query tile (of 64)
  const int h  = blockIdx.y;   // head
  const int b  = blockIdx.z;   // batch

  const float* base = qkv + (size_t)b * NTOK * C3;
  const float scale = 0.125f;  // 1/sqrt(64)

  // Q fragments (A layout), scale folded in. Row = qt*16 + l15.
  const float* qrow = base + (size_t)(qt * 16 + l15) * C3 + h * HD;
  v16bf qf0 = load_a_frag(qrow + 0, half, scale);
  v16bf qf1 = load_a_frag(qrow + 32, half, scale);

  // S = Q K^T, tile by tile into LDS.
  for (int kt = 0; kt < NTOK / 16; ++kt) {
    const float* krow = base + (size_t)(kt * 16 + l15) * C3 + DMODEL + h * HD;
    v16bf kf0 = load_b_fragT(krow + 0, half);   // B[k][n] = K[n][k]
    v16bf kf1 = load_b_fragT(krow + 32, half);
    v8f acc = vzero8();
    acc = wmma_bf16(qf0, kf0, acc);
    acc = wmma_bf16(qf1, kf1, acc);
#pragma unroll
    for (int r = 0; r < 8; ++r)
      S[(r + 8 * half) * NTOK + kt * 16 + l15] = acc[r];
  }
  __syncthreads();

  // Softmax: row = l15, two lanes per row (each scans 512 cols as float4),
  // halves combined via shuffle. Ordered compares (no NaN bookkeeping) so the
  // reductions stay branch-free v_max/v_add loops.
  {
    float4* Sr = (float4*)&S[l15 * NTOK + half * 512];
    float mx = -3.4e38f;
    for (int j = 0; j < 128; ++j) {
      float4 v = Sr[j];
      mx = (v.x > mx) ? v.x : mx;
      mx = (v.y > mx) ? v.y : mx;
      mx = (v.z > mx) ? v.z : mx;
      mx = (v.w > mx) ? v.w : mx;
    }
    const float omx = __shfl_xor(mx, 16, 32);
    mx = (omx > mx) ? omx : mx;
    float sum = 0.0f;
    for (int j = 0; j < 128; ++j) {
      float4 v = Sr[j];
      v.x = __expf(v.x - mx);
      v.y = __expf(v.y - mx);
      v.z = __expf(v.z - mx);
      v.w = __expf(v.w - mx);
      sum += (v.x + v.y) + (v.z + v.w);
      Sr[j] = v;
    }
    sum += __shfl_xor(sum, 16, 32);
    const float inv = 1.0f / sum;
    for (int j = 0; j < 128; ++j) {
      float4 v = Sr[j];
      v.x *= inv; v.y *= inv; v.z *= inv; v.w *= inv;
      Sr[j] = v;
    }
  }
  __syncthreads();

  // O = P V : (16 x 1024) @ (1024 x 64), K chunks of 32, 4 n-tiles of 16.
  v8f o0 = vzero8(), o1 = vzero8(), o2 = vzero8(), o3 = vzero8();
  const float* vbase = base + 2 * DMODEL + h * HD;

#if HAVE_TDM
  auto av_step = [&](int kc, const float* __restrict__ VL) {
    v16bf pf;
#pragma unroll
    for (int e = 0; e < 16; ++e) {
      int k = kc * 32 + half * 16 + e;
      pf[e] = (__bf16)S[l15 * NTOK + k];
    }
    v16bf vf0, vf1, vf2, vf3;
#pragma unroll
    for (int e = 0; e < 16; ++e) {
      const float* vr = VL + (size_t)(half * 16 + e) * HD + l15;
      vf0[e] = (__bf16)vr[0];
      vf1[e] = (__bf16)vr[16];
      vf2[e] = (__bf16)vr[32];
      vf3[e] = (__bf16)vr[48];
    }
    o0 = wmma_bf16(pf, vf0, o0);
    o1 = wmma_bf16(pf, vf1, o1);
    o2 = wmma_bf16(pf, vf2, o2);
    o3 = wmma_bf16(pf, vf3, o3);
  };

  tdm_load_2d_f32(&Vt[0][0], vbase, HD, 32, C3);
  for (int kc = 0; kc < NTOK / 32 - 1; ++kc) {
    // drain LDS reads of the buffer we are about to overwrite, then prefetch
    asm volatile("s_wait_dscnt 0x0" ::: "memory");
    tdm_load_2d_f32(&Vt[(kc + 1) & 1][0], vbase + (size_t)((kc + 1) * 32) * C3, HD, 32, C3);
    __builtin_amdgcn_s_wait_tensorcnt(1);   // current tile resident
    av_step(kc, &Vt[kc & 1][0]);
  }
  __builtin_amdgcn_s_wait_tensorcnt(0);
  av_step(NTOK / 32 - 1, &Vt[(NTOK / 32 - 1) & 1][0]);
#else
  for (int kc = 0; kc < NTOK / 32; ++kc) {
    v16bf pf;
#pragma unroll
    for (int e = 0; e < 16; ++e) {
      int k = kc * 32 + half * 16 + e;
      pf[e] = (__bf16)S[l15 * NTOK + k];
    }
    const float* vcol = vbase + (size_t)(kc * 32) * C3 + l15;
    o0 = wmma_bf16(pf, load_b_frag_strided(vcol + 0,  C3, half), o0);
    o1 = wmma_bf16(pf, load_b_frag_strided(vcol + 16, C3, half), o1);
    o2 = wmma_bf16(pf, load_b_frag_strided(vcol + 32, C3, half), o2);
    o3 = wmma_bf16(pf, load_b_frag_strided(vcol + 48, C3, half), o3);
  }
#endif

  v8f os[4] = {o0, o1, o2, o3};
#pragma unroll
  for (int nt = 0; nt < 4; ++nt) {
#pragma unroll
    for (int r = 0; r < 8; ++r) {
      out[((size_t)b * NTOK + qt * 16 + r + 8 * half) * DMODEL + h * HD + nt * 16 + l15] =
          os[nt][r];
    }
  }
}

// ---------------------------------------------------------------------------
// LayerNorm over last dim (768). One row per block of 256.
// ---------------------------------------------------------------------------
__global__ __launch_bounds__(256) void ln_kernel(const float* __restrict__ x,
                                                 const float* __restrict__ g,
                                                 const float* __restrict__ bb,
                                                 float* __restrict__ out) {
  __shared__ float red[256];
  const int t = threadIdx.x;
  const float* xr = x + (size_t)blockIdx.x * 768;
  float v0 = xr[t], v1 = xr[t + 256], v2 = xr[t + 512];
  red[t] = v0 + v1 + v2;
  __syncthreads();
  for (int s = 128; s > 0; s >>= 1) {
    if (t < s) red[t] += red[t + s];
    __syncthreads();
  }
  const float mu = red[0] * (1.0f / 768.0f);
  __syncthreads();
  const float d0 = v0 - mu, d1 = v1 - mu, d2 = v2 - mu;
  red[t] = d0 * d0 + d1 * d1 + d2 * d2;
  __syncthreads();
  for (int s = 128; s > 0; s >>= 1) {
    if (t < s) red[t] += red[t + s];
    __syncthreads();
  }
  const float rstd = rsqrtf(red[0] * (1.0f / 768.0f) + 1e-5f);
  float* o = out + (size_t)blockIdx.x * 768;
  o[t]       = d0 * rstd * g[t]       + bb[t];
  o[t + 256] = d1 * rstd * g[t + 256] + bb[t + 256];
  o[t + 512] = d2 * rstd * g[t + 512] + bb[t + 512];
}

// L2-normalize rows of length 768.
__global__ __launch_bounds__(256) void rownorm_kernel(const float* __restrict__ in,
                                                      float* __restrict__ out) {
  __shared__ float red[256];
  const int t = threadIdx.x;
  const float* xr = in + (size_t)blockIdx.x * 768;
  float v0 = xr[t], v1 = xr[t + 256], v2 = xr[t + 512];
  red[t] = v0 * v0 + v1 * v1 + v2 * v2;
  __syncthreads();
  for (int s = 128; s > 0; s >>= 1) {
    if (t < s) red[t] += red[t + s];
    __syncthreads();
  }
  const float inv = rsqrtf(red[0] + 1e-24f);
  float* o = out + (size_t)blockIdx.x * 768;
  o[t] = v0 * inv;
  o[t + 256] = v1 * inv;
  o[t + 512] = v2 * inv;
}

__global__ __launch_bounds__(256) void add3_kernel(const float* __restrict__ a,
                                                   const float* __restrict__ b,
                                                   const float* __restrict__ c,
                                                   float* __restrict__ o, int n) {
  int i = blockIdx.x * 256 + threadIdx.x;
  if (i < n) o[i] = a[i] + b[i] + c[i];
}

__global__ __launch_bounds__(256) void add2_kernel(const float* __restrict__ a,
                                                   const float* __restrict__ b,
                                                   float* __restrict__ o, int n) {
  int i = blockIdx.x * 256 + threadIdx.x;
  if (i < n) o[i] = a[i] + b[i];
}

// ---------------------------------------------------------------------------
extern "C" void kernel_launch(void* const* d_in, const int* in_sizes, int n_in,
                              void* d_out, int out_size, void* d_ws, size_t ws_size,
                              hipStream_t stream) {
  (void)in_sizes; (void)n_in; (void)out_size; (void)ws_size;

  const float* x        = (const float*)d_in[0];
  const float* ln1_g    = (const float*)d_in[1];
  const float* ln1_b    = (const float*)d_in[2];
  const float* qkv_w    = (const float*)d_in[3];
  const float* qkv_b    = (const float*)d_in[4];
  const float* proj_w   = (const float*)d_in[5];
  const float* proj_b   = (const float*)d_in[6];
  const float* c_qkv_w  = (const float*)d_in[7];
  const float* c_qkv_b  = (const float*)d_in[8];
  const float* c_proj_w = (const float*)d_in[9];
  const float* c_proj_b = (const float*)d_in[10];
  const float* cp_fc1_w = (const float*)d_in[11];
  const float* cp_fc1_b = (const float*)d_in[12];
  const float* cp_fc2_w = (const float*)d_in[13];
  const float* cp_fc2_b = (const float*)d_in[14];
  const float* P        = (const float*)d_in[15];
  const float* ln2_g    = (const float*)d_in[16];
  const float* ln2_b    = (const float*)d_in[17];
  const float* fc1_w    = (const float*)d_in[18];
  const float* fc1_b    = (const float*)d_in[19];
  const float* fc2_w    = (const float*)d_in[20];
  const float* fc2_b    = (const float*)d_in[21];

  const size_t S = 8192;  // 8 * 1024 tokens
  float* ws = (float*)d_ws;
  float* A1 = ws;               // S*768  : n1 -> Un -> zin -> n2
  float* Q1 = ws + S * 768;     // S*2304 : qkv1 -> qkv2 -> z -> mlp hidden (with AT)
  float* AT = ws + S * 3072;    // S*768  : attention outputs (pre-proj)
  float* Y  = ws + S * 3840;    // S*768  : y, later mlp output
  float* XR = ws + S * 4608;    // S*768  : x + y + z
  float* PB = ws + S * 5376;    // 64*768 : normalized parts
  float* C1 = PB + 64 * 768;    // S*64   : cp hidden
  float* Z  = Q1;               // S*768  (qkv2 dead by then)
  float* H  = Q1;               // S*3072 (spans Q1+AT, both dead by then)

  float* xout = (float*)d_out;
  float* dmap = xout + S * 768;

  const int n_tok = (int)(S * 768);

  // n1 = LN1(x)
  ln_kernel<<<(int)S, 256, 0, stream>>>(x, ln1_g, ln1_b, A1);
  // qkv1 = n1 @ qkv_w^T + qkv_b
  gemm_bf16_kernel<8><<<dim3(18, 128), 128, 0, stream>>>(A1, qkv_w, qkv_b, Q1, (int)S, 2304, 768, 0);
  // Un = normalize(n1)  (in place)
  rownorm_kernel<<<(int)S, 256, 0, stream>>>(A1, A1);
  // Pn = normalize(P)
  rownorm_kernel<<<64, 256, 0, stream>>>(P, PB);
  // dmap = Un @ Pn^T   (also the 2nd output)
  gemm_bf16_kernel<4><<<dim3(1, 128), 128, 0, stream>>>(A1, PB, nullptr, dmap, (int)S, 64, 768, 0);
  // cp hidden = gelu(dmap @ cp_fc1^T + b)
  gemm_bf16_kernel<4><<<dim3(1, 128), 128, 0, stream>>>(dmap, cp_fc1_w, cp_fc1_b, C1, (int)S, 64, 64, 1);
  // zin = cp hidden @ cp_fc2^T + b   (overwrites Un)
  gemm_bf16_kernel<8><<<dim3(6, 128), 128, 0, stream>>>(C1, cp_fc2_w, cp_fc2_b, A1, (int)S, 768, 64, 0);
  // attention 1
  attn_kernel<<<dim3(64, 12, 8), 32, 0, stream>>>(Q1, AT);
  // y = att1 @ proj^T + b
  gemm_bf16_kernel<8><<<dim3(6, 128), 128, 0, stream>>>(AT, proj_w, proj_b, Y, (int)S, 768, 768, 0);
  // qkv2 = zin @ c_qkv^T + b
  gemm_bf16_kernel<8><<<dim3(18, 128), 128, 0, stream>>>(A1, c_qkv_w, c_qkv_b, Q1, (int)S, 2304, 768, 0);
  // attention 2
  attn_kernel<<<dim3(64, 12, 8), 32, 0, stream>>>(Q1, AT);
  // z = att2 @ c_proj^T + b
  gemm_bf16_kernel<8><<<dim3(6, 128), 128, 0, stream>>>(AT, c_proj_w, c_proj_b, Z, (int)S, 768, 768, 0);
  // xr = x + y + z
  add3_kernel<<<n_tok / 256, 256, 0, stream>>>(x, Y, Z, XR, n_tok);
  // n2 = LN2(xr)
  ln_kernel<<<(int)S, 256, 0, stream>>>(XR, ln2_g, ln2_b, A1);
  // h = gelu(n2 @ fc1^T + b)
  gemm_bf16_kernel<8><<<dim3(24, 128), 128, 0, stream>>>(A1, fc1_w, fc1_b, H, (int)S, 3072, 768, 1);
  // m = h @ fc2^T + b
  gemm_bf16_kernel<8><<<dim3(6, 128), 128, 0, stream>>>(H, fc2_w, fc2_b, Y, (int)S, 768, 3072, 0);
  // out = xr + m
  add2_kernel<<<n_tok / 256, 256, 0, stream>>>(XR, Y, xout, n_tok);
}
